// PerformerEncoder_36670430773881
// MI455X (gfx1250) — matrix operand order repaired and checked by reading, not compiled, not run on hardware
//
#include <hip/hip_runtime.h>
#include <hip/hip_bf16.h>
#include <math.h>

#define EMBD  512
#define HEADS 8
#define HDIM  64
#define MLPD  2048
#define NLAYER 6
#define KEPS  1e-3f
#define LNEPS 1e-6f

typedef unsigned short bf16r;
typedef __attribute__((ext_vector_type(16))) __bf16 v16bf;
typedef __attribute__((ext_vector_type(8)))  float  v8f;
typedef int v4i __attribute__((vector_size(16)));   // matches async-LDS builtin pointee

#define AS1 __attribute__((address_space(1)))
#define AS3 __attribute__((address_space(3)))

#if defined(__has_builtin)
#if __has_builtin(__builtin_amdgcn_global_load_async_to_lds_b128)
#define HAVE_ASYNC 1
#endif
#endif
#ifndef HAVE_ASYNC
#define HAVE_ASYNC 0
#endif

__device__ __forceinline__ bf16r f2bf(float f) {
  union { float f; unsigned u; } c; c.f = f;
  unsigned u = c.u;
  return (bf16r)((u + 0x7FFFu + ((u >> 16) & 1u)) >> 16);
}
__device__ __forceinline__ float bf2f(bf16r s) {
  union { unsigned u; float f; } c; c.u = ((unsigned)s) << 16; return c.f;
}

__device__ __forceinline__ void wait_async() {
#if HAVE_ASYNC
#if __has_builtin(__builtin_amdgcn_s_wait_asynccnt)
  __builtin_amdgcn_s_wait_asynccnt(0);
#else
  asm volatile("s_wait_asynccnt 0" ::: "memory");
#endif
#endif
}

// Epilogue modes
#define EPI_BF16            0   // store bf16
#define EPI_RELU_EPS        1   // phi(x)=relu(x)+eps, store bf16
#define EPI_GELU_BIAS       2   // gelu(x+bias), store bf16
#define EPI_RESID_BIAS_F32  3   // x + (bias?) + residual, store f32
#define EPI_F32             4   // store f32
#define EPI_BF16T           5   // store bf16 transposed: C[n*ldC+m]

#define LPAD 72   // LDS row stride for a 64-wide k-chunk (bf16 elems), 144B = 9*16B

// ---------------------------------------------------------------------------
// Stage a 128x64 bf16 tile (rows r0.., k-cols k0..k0+63) into LDS [row][k].
// k-contiguous source -> async global->LDS (b128 x4) when tile is full;
// otherwise vector/scalar gather with zero fill.
// ---------------------------------------------------------------------------
__device__ __forceinline__ void stage_tile(const bf16r* __restrict__ gbase,
                                           bf16r* __restrict__ lds,
                                           int r0, int Rdim, int k0,
                                           long sRow, long sK, int tid)
{
  if (sK == 1) {
    const int r  = tid >> 1;
    const int kg = (tid & 1) * 32;
    const int gr = r0 + r;
    bf16r* dst = &lds[r * LPAD + kg];
#if HAVE_ASYNC
    if (r0 + 127 < Rdim) {
      const bf16r* g = gbase + (long)gr * sRow + (long)(k0 + kg);
      AS1 v4i* gp = (AS1 v4i*)(AS1 void*)g;
      AS3 v4i* lp = (AS3 v4i*)(AS3 void*)dst;
      __builtin_amdgcn_global_load_async_to_lds_b128(gp, lp, 0, 0);
      __builtin_amdgcn_global_load_async_to_lds_b128(gp, lp, 16, 0);
      __builtin_amdgcn_global_load_async_to_lds_b128(gp, lp, 32, 0);
      __builtin_amdgcn_global_load_async_to_lds_b128(gp, lp, 48, 0);
      return;
    }
#endif
    uint4 q0 = make_uint4(0u,0u,0u,0u), q1 = q0, q2 = q0, q3 = q0;
    if (gr < Rdim) {
      const uint4* src = (const uint4*)(gbase + (long)gr * sRow + (long)(k0 + kg));
      q0 = src[0]; q1 = src[1]; q2 = src[2]; q3 = src[3];
    }
    uint4* d = (uint4*)dst;
    d[0] = q0; d[1] = q1; d[2] = q2; d[3] = q3;
  } else {
    for (int i = 0; i < 32; ++i) {
      const int lin = tid * 32 + i;
      const int r = lin >> 6, k = lin & 63;
      const int gr = r0 + r;
      bf16r vv = 0;
      if (gr < Rdim) vv = gbase[(long)gr * sRow + (long)(k0 + k) * sK];
      lds[r * LPAD + k] = vv;
    }
  }
}

// ---------------------------------------------------------------------------
// Generic batched bf16 WMMA GEMM:  C[M,N] = A[M,K] * Bt[N,K]^T (f32 accumulate)
// Block tile 128x128, K-chunk 64 (2 wmma k-steps). 256 threads = 8 wave32
// waves (4x2); each wave: 32x64 sub-tile = 2x4 v_wmma_f32_16x16x32_bf16 accs.
// DOUBLE-BUFFERED LDS staging: chunk i+1's async global->LDS loads are issued
// right after the (single) barrier and overlap the 16 WMMAs of chunk i.
// ---------------------------------------------------------------------------
template<int MODE>
__global__ __launch_bounds__(256)
void gemm_wmma_bf16(const bf16r* __restrict__ A, const bf16r* __restrict__ Bt,
                    void* __restrict__ Cp,
                    const float* __restrict__ bias, const float* __restrict__ Res,
                    int M, int N, int K,
                    long sAm, long sAk, long sBn, long sBk,
                    long ldC, long ldRes,
                    long oAo, long oAi, long oBo, long oBi,
                    long oCo, long oCi, long oRo, long oRi,
                    int innerCount)
{
  __shared__ bf16r lsA[2][128 * LPAD];
  __shared__ bf16r lsB[2][128 * LPAD];

  const int tid   = threadIdx.x;
  const int wave  = tid >> 5;
  const int lane  = tid & 31;
  const int waveM = wave >> 1;      // 0..3 -> 32 rows
  const int waveN = wave & 1;       // 0..1 -> 64 cols
  const int m0 = blockIdx.y * 128;
  const int n0 = blockIdx.x * 128;

  const int z  = blockIdx.z;
  const int zo = z / innerCount;
  const int zi = z % innerCount;
  const bf16r* Ab = A  + zo * oAo + zi * oAi;
  const bf16r* Bb = Bt + zo * oBo + zi * oBi;
  const long  offC = zo * oCo + zi * oCi;
  const long  offR = zo * oRo + zi * oRi;

  v8f acc[2][4];
  for (int i = 0; i < 2; ++i)
    for (int j = 0; j < 4; ++j)
      for (int e = 0; e < 8; ++e) acc[i][j][e] = 0.0f;

  const int halfo = (lane >> 4) * 8;    // A fragment: k = 16*(i/8) + 8*half + i%8
  const int halfb = (lane >> 4) * 8;    // B fragment: k = 16*(i/8) + 8*half + i%8
  const int l16   = lane & 15;
  const int mrow  = waveM * 32 + l16;

  const int nch = K >> 6;   // all K in this model are multiples of 64

  // prologue: stage chunk 0 into buffer 0
  stage_tile(Ab, lsA[0], m0, M, 0, sAm, sAk, tid);
  stage_tile(Bb, lsB[0], n0, N, 0, sBn, sBk, tid);

  for (int i = 0; i < nch; ++i) {
    wait_async();        // my chunk-i async loads complete
    __syncthreads();     // all waves: chunk i visible; prior readers of buf[(i+1)&1] done

    if (i + 1 < nch) {   // prefetch chunk i+1 while computing chunk i
      stage_tile(Ab, lsA[(i + 1) & 1], m0, M, (i + 1) * 64, sAm, sAk, tid);
      stage_tile(Bb, lsB[(i + 1) & 1], n0, N, (i + 1) * 64, sBn, sBk, tid);
    }

    const bf16r* bufA = lsA[i & 1];
    const bf16r* bufB = lsB[i & 1];
    union Frag { uint4 q[2]; v16bf v; };
#pragma unroll
    for (int ks = 0; ks < 2; ++ks) {
      Frag afrag[2];
#pragma unroll
      for (int mt = 0; mt < 2; ++mt) {
        const bf16r* base = &bufA[(mrow + mt * 16) * LPAD + ks * 32];
        afrag[mt].q[0] = *(const uint4*)(base + halfo);
        afrag[mt].q[1] = *(const uint4*)(base + 16 + halfo);
      }
#pragma unroll
      for (int nt = 0; nt < 4; ++nt) {
        const int nrow = waveN * 64 + nt * 16 + l16;
        Frag bfrag;
        const bf16r* base = &bufB[nrow * LPAD + ks * 32];
        bfrag.q[0] = *(const uint4*)(base + halfb);
        bfrag.q[1] = *(const uint4*)(base + 16 + halfb);
#pragma unroll
        for (int mt = 0; mt < 2; ++mt) {
          acc[mt][nt] = __builtin_amdgcn_wmma_f32_16x16x32_bf16(
              false, afrag[mt].v, false, bfrag.v, (short)0, acc[mt][nt], false, false);
        }
      }
    }
  }

  // ---- epilogue (MODE is compile-time) ----
  const int halfsel = lane >> 4;
#pragma unroll
  for (int mt = 0; mt < 2; ++mt) {
#pragma unroll
    for (int nt = 0; nt < 4; ++nt) {
#pragma unroll
      for (int r = 0; r < 8; ++r) {
        const int m = m0 + waveM * 32 + mt * 16 + halfsel * 8 + r;
        const int n = n0 + waveN * 64 + nt * 16 + l16;
        if (m >= M || n >= N) continue;
        float v = acc[mt][nt][r];
        if (MODE == EPI_BF16) {
          ((bf16r*)Cp)[offC + (long)m * ldC + n] = f2bf(v);
        } else if (MODE == EPI_RELU_EPS) {
          ((bf16r*)Cp)[offC + (long)m * ldC + n] = f2bf(fmaxf(v, 0.0f) + KEPS);
        } else if (MODE == EPI_GELU_BIAS) {
          if (bias) v += bias[n];
          const float g = 0.5f * v *
              (1.0f + tanhf(0.7978845608028654f * (v + 0.044715f * v * v * v)));
          ((bf16r*)Cp)[offC + (long)m * ldC + n] = f2bf(g);
        } else if (MODE == EPI_RESID_BIAS_F32) {
          if (bias) v += bias[n];
          v += Res[offR + (long)m * ldRes + n];
          ((float*)Cp)[offC + (long)m * ldC + n] = v;
        } else if (MODE == EPI_F32) {
          ((float*)Cp)[offC + (long)m * ldC + n] = v;
        } else { // EPI_BF16T
          ((bf16r*)Cp)[offC + (long)n * ldC + m] = f2bf(v);
        }
      }
    }
  }
}

// ---------------------------------------------------------------------------
// Embedding gather + sinusoidal positional encoding (f32 residual stream)
// ---------------------------------------------------------------------------
__global__ __launch_bounds__(256)
void embed_kernel(const int* __restrict__ tokens, const float* __restrict__ emb,
                  float* __restrict__ x, int S)
{
  const long row = blockIdx.x;
  const int  s   = (int)(row % S);
  const int  tok = tokens[row];
  for (int d = threadIdx.x; d < EMBD; d += 256) {
    const int i = d >> 1;
    const float dv = __expf(-logf(10000.0f) * (float)(2 * i) / (float)EMBD);
    const float ang = (float)s * dv;
    const float pe = (d & 1) ? __cosf(ang) : __sinf(ang);
    x[row * EMBD + d] = emb[(long)tok * EMBD + d] + pe;
  }
}

// ---------------------------------------------------------------------------
// LayerNorm over EMBD=512: one wave32 per row, shfl reductions.
// ---------------------------------------------------------------------------
__global__ __launch_bounds__(256)
void layernorm_kernel(const float* __restrict__ x, const float* __restrict__ sc,
                      const float* __restrict__ bi, void* __restrict__ out,
                      int outBf16, long nrows)
{
  const long w = (long)blockIdx.x * 8 + (threadIdx.x >> 5);
  const int lane = threadIdx.x & 31;
  if (w >= nrows) return;
  const float* row = x + w * EMBD;
  float vals[16];
  float sum = 0.0f, sq = 0.0f;
  for (int j = 0; j < 16; ++j) {
    const float v = row[lane + j * 32];
    vals[j] = v; sum += v; sq += v * v;
  }
  for (int off = 16; off >= 1; off >>= 1) {
    sum += __shfl_xor(sum, off, 32);
    sq  += __shfl_xor(sq,  off, 32);
  }
  const float mu  = sum * (1.0f / EMBD);
  const float var = sq * (1.0f / EMBD) - mu * mu;
  const float inv = rsqrtf(var + LNEPS);
  for (int j = 0; j < 16; ++j) {
    const int d = lane + j * 32;
    const float y = (vals[j] - mu) * inv * sc[d] + bi[d];
    if (outBf16) ((bf16r*)out)[w * EMBD + d] = f2bf(y);
    else         ((float*)out)[w * EMBD + d] = y;
  }
}

// ---------------------------------------------------------------------------
// ksum[b,h,f] = sum_s pk[b,s,h,f]   (atomicAdd over S chunks; pre-zeroed)
// ---------------------------------------------------------------------------
__global__ __launch_bounds__(64)
void ksum_kernel(const bf16r* __restrict__ pk, float* __restrict__ ksum,
                 int S, int chunk)
{
  const int z = blockIdx.x;
  const int c = blockIdx.y;
  const int b = z / HEADS, h = z % HEADS;
  const int f = threadIdx.x;
  const long base = (long)b * S * EMBD + (long)h * HDIM + f;
  float acc = 0.0f;
  const int s0 = c * chunk;
  const int s1 = min(S, s0 + chunk);
  for (int sI = s0; sI < s1; ++sI) acc += bf2f(pk[base + (long)sI * EMBD]);
  atomicAdd(&ksum[z * HDIM + f], acc);
}

// ---------------------------------------------------------------------------
// attn = num / (pq . ksum) per (b,s,h); one wave per (b,s,h); bf16 output.
// ---------------------------------------------------------------------------
__global__ __launch_bounds__(256)
void attn_kernel(const float* __restrict__ num, const bf16r* __restrict__ pq,
                 const float* __restrict__ ksum, bf16r* __restrict__ attn,
                 int S, long total)
{
  const long w = (long)blockIdx.x * 8 + (threadIdx.x >> 5);
  const int lane = threadIdx.x & 31;
  if (w >= total) return;
  const int  h  = (int)(w % HEADS);
  const long bs = w / HEADS;
  const int  b  = (int)(bs / S);
  const long base = bs * EMBD + (long)h * HDIM;
  const float* ks = ksum + (long)(b * HEADS + h) * HDIM;
  float p = bf2f(pq[base + lane])      * ks[lane]
          + bf2f(pq[base + 32 + lane]) * ks[32 + lane];
  for (int off = 16; off >= 1; off >>= 1) p += __shfl_xor(p, off, 32);
  const float invden = 1.0f / p;
  attn[base + lane]      = f2bf(num[base + lane]      * invden);
  attn[base + 32 + lane] = f2bf(num[base + 32 + lane] * invden);
}

// ---------------------------------------------------------------------------
// f32 [L][K][N] -> bf16 [L][N][K] transposed conversion (32x32 LDS tiles)
// ---------------------------------------------------------------------------
__global__ __launch_bounds__(256)
void f2bfT_kernel(const float* __restrict__ src, bf16r* __restrict__ dst,
                  int K, int N)
{
  __shared__ float tile[32][33];
  const long zoff = (long)blockIdx.z * K * N;
  const int kb = blockIdx.y * 32, nb = blockIdx.x * 32;
  const int tx = threadIdx.x & 31, ty = threadIdx.x >> 5;
  for (int i = 0; i < 32; i += 8)
    tile[ty + i][tx] = src[zoff + (long)(kb + ty + i) * N + (nb + tx)];
  __syncthreads();
  for (int i = 0; i < 32; i += 8)
    dst[zoff + (long)(nb + ty + i) * K + (kb + tx)] = f2bf(tile[tx][ty + i]);
}

// ---------------------------------------------------------------------------
// Host-side orchestration
// ---------------------------------------------------------------------------
static void launch_gemm(hipStream_t stream, const bf16r* A, const bf16r* Bt, void* C,
                        const float* bias, const float* res,
                        int M, int N, int K,
                        long sAm, long sAk, long sBn, long sBk,
                        long ldC, long ldRes,
                        int batches, int innerCount,
                        long oAo, long oAi, long oBo, long oBi,
                        long oCo, long oCi, long oRo, long oRi,
                        int mode)
{
  dim3 grid((N + 127) / 128, (M + 127) / 128, batches);
#define GEMM_CASE(MD) gemm_wmma_bf16<MD><<<grid, 256, 0, stream>>>( \
      A, Bt, C, bias, res, M, N, K, sAm, sAk, sBn, sBk, ldC, ldRes, \
      oAo, oAi, oBo, oBi, oCo, oCi, oRo, oRi, innerCount)
  switch (mode) {
    case EPI_BF16:           GEMM_CASE(EPI_BF16);           break;
    case EPI_RELU_EPS:       GEMM_CASE(EPI_RELU_EPS);       break;
    case EPI_GELU_BIAS:      GEMM_CASE(EPI_GELU_BIAS);      break;
    case EPI_RESID_BIAS_F32: GEMM_CASE(EPI_RESID_BIAS_F32); break;
    case EPI_F32:            GEMM_CASE(EPI_F32);            break;
    default:                 GEMM_CASE(EPI_BF16T);          break;
  }
#undef GEMM_CASE
}

extern "C" void kernel_launch(void* const* d_in, const int* in_sizes, int n_in,
                              void* d_out, int out_size, void* d_ws, size_t ws_size,
                              hipStream_t stream)
{
  const int Bsz = 4, S = 4096;
  const long R = (long)Bsz * S;               // 16384 rows

  const int*   tokens = (const int*)  d_in[0];
  const float* embedT = (const float*)d_in[1];
  const float* Wq     = (const float*)d_in[2];
  const float* Wk     = (const float*)d_in[3];
  const float* Wv     = (const float*)d_in[4];
  const float* Wo     = (const float*)d_in[5];
  const float* ln1_s  = (const float*)d_in[6];
  const float* ln1_b  = (const float*)d_in[7];
  const float* W1     = (const float*)d_in[8];
  const float* b1     = (const float*)d_in[9];
  const float* W2     = (const float*)d_in[10];
  const float* b2     = (const float*)d_in[11];
  const float* ln2_s  = (const float*)d_in[12];
  const float* ln2_b  = (const float*)d_in[13];
  const float* lnf_s  = (const float*)d_in[14];
  const float* lnf_b  = (const float*)d_in[15];

  // ---- workspace arena ----
  char* wsp = (char*)d_ws;
  auto alloc = [&](size_t bytes) -> char* {
    char* p = wsp;
    wsp += (bytes + 255) & ~(size_t)255;
    return p;
  };
  float* x      = (float*)alloc(R * EMBD * 4);
  bf16r* hbuf   = (bf16r*)alloc(R * EMBD * 2);
  bf16r* pq     = (bf16r*)alloc(R * EMBD * 2);
  bf16r* pk     = (bf16r*)alloc(R * EMBD * 2);
  bf16r* vbuf   = (bf16r*)alloc(R * EMBD * 2);
  bf16r* attn   = (bf16r*)alloc(R * EMBD * 2);
  bf16r* mlpbuf = (bf16r*)alloc(R * MLPD * 2);
  float* numb   = (float*)alloc(R * EMBD * 4);
  bf16r* kvT    = (bf16r*)alloc((long)Bsz * HEADS * HDIM * HDIM * 2);
  float* ksum   = (float*)alloc((long)Bsz * HEADS * HDIM * 4);
  bf16r* WqT    = (bf16r*)alloc((long)NLAYER * EMBD * EMBD * 2);
  bf16r* WkT    = (bf16r*)alloc((long)NLAYER * EMBD * EMBD * 2);
  bf16r* WvT    = (bf16r*)alloc((long)NLAYER * EMBD * EMBD * 2);
  bf16r* WoT    = (bf16r*)alloc((long)NLAYER * EMBD * EMBD * 2);
  bf16r* W1T    = (bf16r*)alloc((long)NLAYER * EMBD * MLPD * 2);
  bf16r* W2T    = (bf16r*)alloc((long)NLAYER * MLPD * EMBD * 2);

  // ---- one-time weight conversion f32 -> bf16, TRANSPOSED to [N][K] ----
  f2bfT_kernel<<<dim3(EMBD/32, EMBD/32, NLAYER), 256, 0, stream>>>(Wq, WqT, EMBD, EMBD);
  f2bfT_kernel<<<dim3(EMBD/32, EMBD/32, NLAYER), 256, 0, stream>>>(Wk, WkT, EMBD, EMBD);
  f2bfT_kernel<<<dim3(EMBD/32, EMBD/32, NLAYER), 256, 0, stream>>>(Wv, WvT, EMBD, EMBD);
  f2bfT_kernel<<<dim3(EMBD/32, EMBD/32, NLAYER), 256, 0, stream>>>(Wo, WoT, EMBD, EMBD);
  f2bfT_kernel<<<dim3(MLPD/32, EMBD/32, NLAYER), 256, 0, stream>>>(W1, W1T, EMBD, MLPD);
  f2bfT_kernel<<<dim3(EMBD/32, MLPD/32, NLAYER), 256, 0, stream>>>(W2, W2T, MLPD, EMBD);

  // ---- embedding + positional encoding ----
  embed_kernel<<<(unsigned)R, 256, 0, stream>>>(tokens, embedT, x, S);

  const long SE = (long)S * EMBD;
  const int  BH = Bsz * HEADS;
  const long WS = (long)EMBD * EMBD;   // per-layer square weight stride

  for (int l = 0; l < NLAYER; ++l) {
    // h = LN1(x), bf16
    layernorm_kernel<<<(unsigned)(R / 8), 256, 0, stream>>>(
        x, ln1_s + (long)l * EMBD, ln1_b + (long)l * EMBD, hbuf, 1, R);

    // pq = relu(h Wq)+eps ; pk = relu(h Wk)+eps ; v = h Wv   (Bt = [N][K])
    launch_gemm(stream, hbuf, WqT + l * WS, pq, nullptr, nullptr, (int)R, EMBD, EMBD,
                EMBD, 1, EMBD, 1, EMBD, 0, 1, 1, 0,0,0,0,0,0,0,0, EPI_RELU_EPS);
    launch_gemm(stream, hbuf, WkT + l * WS, pk, nullptr, nullptr, (int)R, EMBD, EMBD,
                EMBD, 1, EMBD, 1, EMBD, 0, 1, 1, 0,0,0,0,0,0,0,0, EPI_RELU_EPS);
    launch_gemm(stream, hbuf, WvT + l * WS, vbuf, nullptr, nullptr, (int)R, EMBD, EMBD,
                EMBD, 1, EMBD, 1, EMBD, 0, 1, 1, 0,0,0,0,0,0,0,0, EPI_BF16);

    // ksum[b,h,f] = sum_s pk
    (void)hipMemsetAsync(ksum, 0, (size_t)BH * HDIM * 4, stream);
    ksum_kernel<<<dim3(BH, 32), 64, 0, stream>>>(pk, ksum, S, S / 32);

    // kvT[b,h][e][f] = (pk^T @ v)^T : batched over (b,h); both operands
    // S-strided -> scalar staging path; transposed store feeds num GEMM.
    launch_gemm(stream, pk, vbuf, kvT, nullptr, nullptr, HDIM, HDIM, S,
                1, EMBD,            // A: (f,s) -> pk + f + s*512
                1, EMBD,            // Bt: (e,s) -> v + e + s*512
                HDIM, 0, BH, HEADS,
                SE, HDIM, SE, HDIM,
                (long)HEADS * HDIM * HDIM, (long)HDIM * HDIM,
                0, 0, EPI_BF16T);

    // num = pq @ kv : batched over (b,h); Bt = kvT [e][f] k-contiguous
    launch_gemm(stream, pq, kvT, numb, nullptr, nullptr, S, HDIM, HDIM,
                EMBD, 1,            // A: (s,f) -> pq + s*512 + f
                HDIM, 1,            // Bt: (e,f) -> kvT + e*64 + f
                EMBD, 0, BH, HEADS,
                SE, HDIM,
                (long)HEADS * HDIM * HDIM, (long)HDIM * HDIM,
                SE, HDIM,
                0, 0, EPI_F32);

    // attn = num / (pq . ksum), bf16
    attn_kernel<<<(unsigned)(R * HEADS / 8), 256, 0, stream>>>(
        numb, pq, ksum, attn, S, R * HEADS);

    // x = x + attn @ Wo  (f32 residual)
    launch_gemm(stream, attn, WoT + l * WS, x, nullptr, x, (int)R, EMBD, EMBD,
                EMBD, 1, EMBD, 1, EMBD, EMBD, 1, 1,
                0,0,0,0,0,0,0,0, EPI_RESID_BIAS_F32);

    // h = LN2(x), bf16
    layernorm_kernel<<<(unsigned)(R / 8), 256, 0, stream>>>(
        x, ln2_s + (long)l * EMBD, ln2_b + (long)l * EMBD, hbuf, 1, R);

    // mlp = gelu(h W1 + b1), bf16 ; W1T rows are N=MLPD, K=EMBD
    launch_gemm(stream, hbuf, W1T + (long)l * EMBD * MLPD, mlpbuf,
                b1 + (long)l * MLPD, nullptr,
                (int)R, MLPD, EMBD, EMBD, 1, EMBD, 1, MLPD, 0, 1, 1,
                0,0,0,0,0,0,0,0, EPI_GELU_BIAS);

    // x = x + mlp W2 + b2 ; W2T rows are N=EMBD, K=MLPD
    launch_gemm(stream, mlpbuf, W2T + (long)l * MLPD * EMBD, x,
                b2 + (long)l * EMBD, x,
                (int)R, EMBD, MLPD, MLPD, 1, MLPD, 1, EMBD, EMBD, 1, 1,
                0,0,0,0,0,0,0,0, EPI_RESID_BIAS_F32);
  }

  // final LayerNorm -> f32 output
  layernorm_kernel<<<(unsigned)(R / 8), 256, 0, stream>>>(
      x, lnf_s, lnf_b, d_out, 0, R);
}
